// BiologicalAttention_37022618091684
// MI455X (gfx1250) — matrix operand order repaired and checked by reading, not compiled, not run on hardware
//
#include <hip/hip_runtime.h>
#include <hip/hip_bf16.h>

// ---------------------------------------------------------------------------
// BiologicalAttention fused implementation for gfx1250 (MI455X, wave32, WMMA)
//   B=4, N=2048, C=256, H=8, D=32
// Strategy:
//   prep:      x -> bf16, W_qkv -> W_qkv^T bf16, W_out -> W_out^T bf16
//   qkv_gemm:  WMMA bf16 GEMM, writes Q[B,H,N,D], K[B,H,N,D], V^T[B,H,D,N]
//   attn:      fused flash attention (S^T = K*Q^T trick so the score tile's
//              C-layout is directly the A-layout of the P*V WMMA)
//   out_gemm:  WMMA bf16 GEMM + bias -> fp32 d_out
// ---------------------------------------------------------------------------

typedef __attribute__((ext_vector_type(16))) __bf16 bf16x16;
typedef __attribute__((ext_vector_type(8)))  float  f32x8;

#define NSEQ 2048
#define NBH  32          // B*H
#define CDIM 256

union FragU { uint4 u[2]; bf16x16 v; };

__device__ __forceinline__ bf16x16 ld_frag(const __bf16* p0, const __bf16* p1) {
  FragU f;
  f.u[0] = *reinterpret_cast<const uint4*>(p0);
  f.u[1] = *reinterpret_cast<const uint4*>(p1);
  return f.v;
}

// 16x16 f32 tile of (A row-major [*,256]) x (B^T row-major [*,256]), K=256.
// arow = A + row_base*256 (row of this lane), brow = B^T + col_base*256.
__device__ __forceinline__ f32x8 gemm_k256(const __bf16* arow, const __bf16* brow,
                                           int hb) {
  f32x8 acc = {};
#pragma unroll
  for (int kb = 0; kb < 256; kb += 32) {
    // A frag: elems 0..7 -> K = kb + hb*8 .. +7 ; elems 8..15 -> +16
    bf16x16 a = ld_frag(arow + kb + hb * 8, arow + kb + hb * 8 + 16);
    // B frag: elems 0..7 -> K = kb + hb*8 .. +7 ; elems 8..15 -> +16
    bf16x16 b = ld_frag(brow + kb + hb * 8, brow + kb + hb * 8 + 16);
    acc = __builtin_amdgcn_wmma_f32_16x16x32_bf16(false, a, false, b,
                                                  (short)0, acc, false, false);
  }
  return acc;
}

// ---------------------------------------------------------------------------
// Kernel 0: dtype conversion + weight transposes
// ---------------------------------------------------------------------------
__global__ __launch_bounds__(256) void prep_kernel(
    const float* __restrict__ x, const float* __restrict__ wqkv,
    const float* __restrict__ wout, __bf16* __restrict__ xb,
    __bf16* __restrict__ wqkvT, __bf16* __restrict__ woutT) {
  const int tid = blockIdx.x * blockDim.x + threadIdx.x;
  const int stride = gridDim.x * blockDim.x;
  for (int j = tid; j < 8192 * 256; j += stride) xb[j] = (__bf16)x[j];
  for (int j = tid; j < 256 * 768; j += stride) {
    int k = j / 768, n = j % 768;
    wqkvT[n * 256 + k] = (__bf16)wqkv[j];
  }
  for (int j = tid; j < 256 * 256; j += stride) {
    int k = j / 256, n = j % 256;
    woutT[n * 256 + k] = (__bf16)wout[j];
  }
}

// ---------------------------------------------------------------------------
// Kernel 1: QKV projection.  tiles: M = 8192/16 = 512, Nout = 768/16 = 48
// One wave per 16x16 tile, 8 waves per block -> 3072 blocks.
// ---------------------------------------------------------------------------
__global__ __launch_bounds__(256) void qkv_gemm_kernel(
    const __bf16* __restrict__ xb, const __bf16* __restrict__ wT,
    const float* __restrict__ bias, __bf16* __restrict__ Qb,
    __bf16* __restrict__ Kb, __bf16* __restrict__ Vt) {
  const int wave = threadIdx.x >> 5, lane = threadIdx.x & 31;
  const int hb = lane >> 4, ln = lane & 15;
  const int tile = blockIdx.x * 8 + wave;
  const int mt = tile / 48, nt = tile % 48;

  const __bf16* arow = xb + (size_t)(mt * 16 + ln) * 256;
  const __bf16* brow = wT + (size_t)(nt * 16 + ln) * 256;
  f32x8 acc = gemm_k256(arow, brow, hb);

  const int c3 = nt * 16 + ln;        // column in [0,768)
  const float bv = bias[c3];
  const int which = c3 >> 8;          // 0=q 1=k 2=v
  const int h = (c3 >> 5) & 7;
  const int d = c3 & 31;
#pragma unroll
  for (int r = 0; r < 8; ++r) {
    const int mrow = mt * 16 + r + hb * 8;
    const int b = mrow >> 11, nn = mrow & 2047;
    const __bf16 v = (__bf16)(acc[r] + bv);
    const size_t bh = (size_t)(b * 8 + h);
    if (which == 0)      Qb[(bh * NSEQ + nn) * 32 + d] = v;
    else if (which == 1) Kb[(bh * NSEQ + nn) * 32 + d] = v;
    else                 Vt[(bh * 32 + d) * NSEQ + nn] = v;
  }
}

// ---------------------------------------------------------------------------
// Kernel 2: fused flash attention.
// grid = 512 blocks (32 bh * 16 q-blocks of 128), 256 threads = 8 waves,
// each wave owns a 16-query tile; K/V^T staged in LDS per 32-key block.
// ---------------------------------------------------------------------------
__global__ __launch_bounds__(256) void attn_kernel(
    const __bf16* __restrict__ Qb, const __bf16* __restrict__ Kb,
    const __bf16* __restrict__ Vt, const float* __restrict__ mask,
    __bf16* __restrict__ O) {
  __shared__ __bf16 kTile[32 * 40];   // [key][d], padded stride 40 (80B)
  __shared__ __bf16 vTile[32 * 40];   // [d][key], padded stride 40

  const int tid = threadIdx.x;
  const int wave = tid >> 5, lane = tid & 31;
  const int hb = lane >> 4, ln = lane & 15;
  const int bh = blockIdx.x >> 4;
  const int qblk = blockIdx.x & 15;
  const int q0 = qblk * 128 + wave * 16;
  const int qg = q0 + ln;             // this lane's query index (score column)

  const __bf16* Qh = Qb + (size_t)bh * NSEQ * 32;
  const __bf16* Kh = Kb + (size_t)bh * NSEQ * 32;
  const __bf16* Vh = Vt + (size_t)bh * 32 * NSEQ;

  // B-fragment of S^T WMMA: Q^T column, d = hb*8..+7 and hb*8+16..+23 of this lane's query
  const bf16x16 qf =
      ld_frag(Qh + (size_t)qg * 32 + hb * 8, Qh + (size_t)qg * 32 + hb * 8 + 16);

  f32x8 acc0 = {}, acc1 = {};
  float m = -__builtin_inff(), l = 0.f;
  const float sc = 0.17677669529663687f;   // 1/sqrt(32)

  for (int kb0 = 0; kb0 < NSEQ; kb0 += 32) {
    __syncthreads();
    {   // cooperative staging: 256 threads x 8B each per tile
      const int row = tid >> 3, c4 = (tid & 7) * 4;
      *reinterpret_cast<uint2*>(&kTile[row * 40 + c4]) =
          *reinterpret_cast<const uint2*>(&Kh[(size_t)(kb0 + row) * 32 + c4]);
      *reinterpret_cast<uint2*>(&vTile[row * 40 + c4]) =
          *reinterpret_cast<const uint2*>(&Vh[(size_t)row * NSEQ + kb0 + c4]);
    }
    __syncthreads();

    // S^T tiles: A = K rows (16 keys x 32 d), B = Q^T
    bf16x16 a0 = ld_frag(&kTile[(ln)      * 40 + hb * 8],
                         &kTile[(ln)      * 40 + hb * 8 + 16]);
    bf16x16 a1 = ld_frag(&kTile[(16 + ln) * 40 + hb * 8],
                         &kTile[(16 + ln) * 40 + hb * 8 + 16]);
    f32x8 z = {};
    f32x8 s0 = __builtin_amdgcn_wmma_f32_16x16x32_bf16(false, a0, false, qf,
                                                       (short)0, z, false, false);
    f32x8 s1 = __builtin_amdgcn_wmma_f32_16x16x32_bf16(false, a1, false, qf,
                                                       (short)0, z, false, false);

    // scale + biological mask (element (q=qg, key = kb0 + tile*16 + r + hb*8))
    float sv[16];
    {
      const float* mrow = mask + (size_t)qg * NSEQ + kb0 + hb * 8;
      const float4 ma = *reinterpret_cast<const float4*>(mrow);
      const float4 mb = *reinterpret_cast<const float4*>(mrow + 4);
      const float4 mc = *reinterpret_cast<const float4*>(mrow + 16);
      const float4 md = *reinterpret_cast<const float4*>(mrow + 20);
      const float mm[16] = {ma.x, ma.y, ma.z, ma.w, mb.x, mb.y, mb.z, mb.w,
                            mc.x, mc.y, mc.z, mc.w, md.x, md.y, md.z, md.w};
#pragma unroll
      for (int i = 0; i < 8; ++i) sv[i]     = s0[i] * sc * (0.1f + 0.9f * mm[i]);
#pragma unroll
      for (int i = 0; i < 8; ++i) sv[8 + i] = s1[i] * sc * (0.1f + 0.9f * mm[8 + i]);
    }

    // online softmax over this lane's 16 keys + the opposite half's 16
    float mloc = sv[0];
#pragma unroll
    for (int i = 1; i < 16; ++i) mloc = fmaxf(mloc, sv[i]);
    const float mblk = fmaxf(mloc, __shfl_xor(mloc, 16, 32));
    const float mnew = fmaxf(m, mblk);
    const float alpha = __expf(m - mnew);
    float rs = 0.f;
#pragma unroll
    for (int i = 0; i < 16; ++i) { sv[i] = __expf(sv[i] - mnew); rs += sv[i]; }
    rs += __shfl_xor(rs, 16, 32);
    l = l * alpha + rs;
    m = mnew;

    // rescale O accumulators (row r of C holds local query r + hb*8)
#pragma unroll
    for (int r = 0; r < 8; ++r) {
      const float ar = __shfl(alpha, r + hb * 8, 32);
      acc0[r] *= ar;
      acc1[r] *= ar;
    }

    // pack P: S^T C-layout is exactly the A-layout of P (16 queries x 32 keys)
    bf16x16 pf;
#pragma unroll
    for (int i = 0; i < 16; ++i) pf[i] = (__bf16)sv[i];

    // P * V : B-fragments are rows of V^T (contiguous keys)
    bf16x16 v0 = ld_frag(&vTile[(ln)      * 40 + hb * 8],
                         &vTile[(ln)      * 40 + hb * 8 + 16]);
    bf16x16 v1 = ld_frag(&vTile[(16 + ln) * 40 + hb * 8],
                         &vTile[(16 + ln) * 40 + hb * 8 + 16]);
    acc0 = __builtin_amdgcn_wmma_f32_16x16x32_bf16(false, pf, false, v0,
                                                   (short)0, acc0, false, false);
    acc1 = __builtin_amdgcn_wmma_f32_16x16x32_bf16(false, pf, false, v1,
                                                   (short)0, acc1, false, false);
  }

  // epilogue: divide by row sums, write bf16 attention output [B,N,C]
  const int b = bh >> 3, h = bh & 7;
  __bf16* obase = O + ((size_t)(b * NSEQ + q0) * CDIM) + h * 32;
#pragma unroll
  for (int r = 0; r < 8; ++r) {
    const int qrow = r + hb * 8;
    const float inv = 1.0f / __shfl(l, qrow, 32);
    obase[(size_t)qrow * CDIM + ln]      = (__bf16)(acc0[r] * inv);
    obase[(size_t)qrow * CDIM + 16 + ln] = (__bf16)(acc1[r] * inv);
  }
}

// ---------------------------------------------------------------------------
// Kernel 3: output projection. tiles: M=512, Nout=256/16=16 -> 1024 blocks
// ---------------------------------------------------------------------------
__global__ __launch_bounds__(256) void out_gemm_kernel(
    const __bf16* __restrict__ attn, const __bf16* __restrict__ wT,
    const float* __restrict__ bias, float* __restrict__ out) {
  const int wave = threadIdx.x >> 5, lane = threadIdx.x & 31;
  const int hb = lane >> 4, ln = lane & 15;
  const int tile = blockIdx.x * 8 + wave;
  const int mt = tile / 16, nt = tile % 16;

  const __bf16* arow = attn + (size_t)(mt * 16 + ln) * 256;
  const __bf16* brow = wT + (size_t)(nt * 16 + ln) * 256;
  f32x8 acc = gemm_k256(arow, brow, hb);

  const int c = nt * 16 + ln;
  const float bv = bias[c];
#pragma unroll
  for (int r = 0; r < 8; ++r) {
    const int mrow = mt * 16 + r + hb * 8;
    out[(size_t)mrow * CDIM + c] = acc[r] + bv;
  }
}

// ---------------------------------------------------------------------------
extern "C" void kernel_launch(void* const* d_in, const int* in_sizes, int n_in,
                              void* d_out, int out_size, void* d_ws, size_t ws_size,
                              hipStream_t stream) {
  const float* x    = (const float*)d_in[0];
  const float* mask = (const float*)d_in[1];
  const float* wqkv = (const float*)d_in[2];
  const float* bqkv = (const float*)d_in[3];
  const float* wout = (const float*)d_in[4];
  const float* bout = (const float*)d_in[5];
  float* out = (float*)d_out;

  char* ws = (char*)d_ws;
  size_t off = 0;
  auto take = [&](size_t bytes) { char* p = ws + off; off += (bytes + 255) & ~(size_t)255; return p; };
  __bf16* xb    = (__bf16*)take((size_t)8192 * 256 * 2);
  __bf16* wqkvT = (__bf16*)take((size_t)768 * 256 * 2);
  __bf16* woutT = (__bf16*)take((size_t)256 * 256 * 2);
  __bf16* Qb    = (__bf16*)take((size_t)NBH * NSEQ * 32 * 2);
  __bf16* Kbuf  = (__bf16*)take((size_t)NBH * NSEQ * 32 * 2);
  __bf16* Vt    = (__bf16*)take((size_t)NBH * 32 * NSEQ * 2);
  __bf16* attn  = (__bf16*)take((size_t)8192 * 256 * 2);

  prep_kernel<<<2048, 256, 0, stream>>>(x, wqkv, wout, xb, wqkvT, woutT);
  qkv_gemm_kernel<<<3072, 256, 0, stream>>>(xb, wqkvT, bqkv, Qb, Kbuf, Vt);
  attn_kernel<<<512, 256, 0, stream>>>(Qb, Kbuf, Vt, mask, attn);
  out_gemm_kernel<<<1024, 256, 0, stream>>>(attn, woutT, bout, out);
}
